// InnerLayer_53798760349843
// MI455X (gfx1250) — matrix operand order repaired and checked by reading, not compiled, not run on hardware
//
#include <hip/hip_runtime.h>

typedef __attribute__((ext_vector_type(16))) _Float16 v16h;
typedef __attribute__((ext_vector_type(2)))  _Float16 h2;
typedef __attribute__((ext_vector_type(8)))  float    v8f;
typedef __attribute__((ext_vector_type(4)))  float    v4f;

union AFrag { v16h v; h2 p[8]; };

// ---------------------------------------------------------------------------
// WMMA fragment layouts (CDNA5, wave32, 16-bit operands):
//  A (16x32, MxK): lane m=lane&15; half-slot j: K = (j>=8?16:0) + (lane>=16?8:0) + (j&7)
//  B (32x16, KxN): lane n=lane&15; half-slot j: K = (j>=8?16:0) + (lane>=16?8:0) + (j&7)
//  C/D (16x16 f32): lane holds N=lane&15; reg r holds M = r + 8*(lane>=16)
// ---------------------------------------------------------------------------

// ============================ edge (NNConv message) ========================
// Fused GEMM: (E x 544) * (544 x 32),  A-row[e] = concat(x_src[e] (x) ea[e], x_src[e])
//   k = f*16 + d  (k < 512):  B[k][g] = msg_W[d][f*32+g]
//   k = 512 + f:              B[k][g] = msg_b[f*32+g]
// Each wave processes TWO 16-edge tiles jointly (shared B fragments).
// Messages scattered into out[dst] with relaxed f32 atomics (segment_sum).
__global__ __launch_bounds__(256) void nnconv_edge_kernel(
    const float* __restrict__ x_src,   // [N_src, 32]
    const int*   __restrict__ ei,      // [2, E] (row0 = src, row1 = dst)
    const float* __restrict__ ea,      // [E, 16]
    const float* __restrict__ msg_W,   // [16, 1024]
    const float* __restrict__ msg_b,   // [1024]
    float*       __restrict__ out,     // [N_dst, 32], accumulated atomically
    int E)
{
    // 17 K-steps x 2 N-tiles = 34 B fragments of 512 halves = 34 KB LDS
    __shared__ __align__(32) _Float16 ldsB[34 * 512];

    const int t = threadIdx.x;

    // ---- Coalesced cooperative fill of B fragments (float4 reads) ----
    // msg_W: 16384 f32 = 4096 float4; element (d,f,g):
    //   frag = (f>>1)*2 + (g>>4); lds = frag*512 + (((d>>3)&1)*16 + (g&15))*16 + (f&1)*8 + (d&7)
    #pragma unroll
    for (int it = 0; it < 16; ++it) {
        const int i  = (it * 256 + t) * 4;          // 4-aligned elem idx (g%4==0)
        const int d  = i >> 10;
        const int f  = (i >> 5) & 31;
        const int g  = i & 31;
        const v4f w  = *(const v4f*)(msg_W + i);
        const int base = ((f >> 1) * 2 + (g >> 4)) * 512
                       + (((d >> 3) & 1) * 16 + (g & 15)) * 16 + (f & 1) * 8 + (d & 7);
        ldsB[base]      = (_Float16)w[0];
        ldsB[base + 16] = (_Float16)w[1];
        ldsB[base + 32] = (_Float16)w[2];
        ldsB[base + 48] = (_Float16)w[3];
    }
    // msg_b: 1024 f32 = 256 float4; element (f,g):
    //   frag = 32 + (g>>4); lds = frag*512 + (((f>>3)&1)*16 + (g&15))*16 + (f>>4)*8 + (f&7)
    {
        const int i  = t * 4;
        const int f  = i >> 5;
        const int g  = i & 31;
        const v4f w  = *(const v4f*)(msg_b + i);
        const int base = (32 + (g >> 4)) * 512
                       + (((f >> 3) & 1) * 16 + (g & 15)) * 16 + (f >> 4) * 8 + (f & 7);
        ldsB[base]      = (_Float16)w[0];
        ldsB[base + 16] = (_Float16)w[1];
        ldsB[base + 32] = (_Float16)w[2];
        ldsB[base + 48] = (_Float16)w[3];
    }
    __syncthreads();

    const int lane = t & 31;
    const int wave = t >> 5;
    const int m    = lane & 15;
    const int hi   = lane >> 4;
    const bool hiF = hi != 0;

    const int e0 = blockIdx.x * 256 + wave * 32;     // two 16-edge tiles per wave
    const int Em1 = E - 1;

    // Per-tile lane state: x row as 16 packed h2 pairs + 4 packed ea pairs.
    h2 xh2[2][16];
    h2 eah2[2][4];
    #pragma unroll
    for (int tl = 0; tl < 2; ++tl) {
        int e = e0 + tl * 16 + m;
        if (e > Em1) e = Em1;                        // clamp loads; EXEC stays full
        const int src = ei[e];
        const v4f* xp4 = (const v4f*)(x_src + (size_t)src * 32);
        #pragma unroll
        for (int q = 0; q < 8; ++q) {
            const v4f xq = xp4[q];
            xh2[tl][2 * q]     = h2{(_Float16)xq[0], (_Float16)xq[1]};
            xh2[tl][2 * q + 1] = h2{(_Float16)xq[2], (_Float16)xq[3]};
        }
        const float* eap = ea + (size_t)e * 16 + hi * 8;
        const v4f ev0 = *(const v4f*)(eap);
        const v4f ev1 = *(const v4f*)(eap + 4);
        eah2[tl][0] = h2{(_Float16)ev0[0], (_Float16)ev0[1]};
        eah2[tl][1] = h2{(_Float16)ev0[2], (_Float16)ev0[3]};
        eah2[tl][2] = h2{(_Float16)ev1[0], (_Float16)ev1[1]};
        eah2[tl][3] = h2{(_Float16)ev1[2], (_Float16)ev1[3]};
    }

    v8f c00 = {}, c01 = {};    // tile0: cols 0..15 / 16..31
    v8f c10 = {}, c11 = {};    // tile1

    #pragma unroll
    for (int s = 0; s < 17; ++s) {
        const v16h b0 = *(const v16h*)&ldsB[(2 * s + 0) * 512 + lane * 16];
        const v16h b1 = *(const v16h*)&ldsB[(2 * s + 1) * 512 + lane * 16];
        AFrag a[2];
        #pragma unroll
        for (int tl = 0; tl < 2; ++tl) {
            if (s < 16) {
                // slots 0-7: x[2s]*ea[d], slots 8-15: x[2s+1]*ea[d]; d = hi*8+(j&7)
                const _Float16 x0 = xh2[tl][s][0];
                const _Float16 x1 = xh2[tl][s][1];
                #pragma unroll
                for (int k = 0; k < 4; ++k) {
                    a[tl].p[k]     = x0 * eah2[tl][k];   // v_pk_mul_f16 (opsel splat)
                    a[tl].p[4 + k] = x1 * eah2[tl][k];
                }
            } else {
                // msg_b step: A[m][f] = x[f]; slots 0-7: f=hi*8+j, 8-15: f=16+hi*8+j
                #pragma unroll
                for (int k = 0; k < 4; ++k) {
                    a[tl].p[k]     = hiF ? xh2[tl][4 + k]  : xh2[tl][k];
                    a[tl].p[4 + k] = hiF ? xh2[tl][12 + k] : xh2[tl][8 + k];
                }
            }
        }
        c00 = __builtin_amdgcn_wmma_f32_16x16x32_f16(false, a[0].v, false, b0, (short)0, c00, false, false);
        c01 = __builtin_amdgcn_wmma_f32_16x16x32_f16(false, a[0].v, false, b1, (short)0, c01, false, false);
        c10 = __builtin_amdgcn_wmma_f32_16x16x32_f16(false, a[1].v, false, b0, (short)0, c10, false, false);
        c11 = __builtin_amdgcn_wmma_f32_16x16x32_f16(false, a[1].v, false, b1, (short)0, c11, false, false);
    }

    // Scatter-add: C row M = r + 8*hi is edge (tile_base + M); lane holds col m, m+16.
    #pragma unroll
    for (int r = 0; r < 8; ++r) {
        const int erow0 = e0 + r + 8 * hi;
        if (erow0 < E) {
            const int dst = ei[E + erow0];
            float* o = out + (size_t)dst * 32 + m;
            __hip_atomic_fetch_add(o,      c00[r], __ATOMIC_RELAXED, __HIP_MEMORY_SCOPE_AGENT);
            __hip_atomic_fetch_add(o + 16, c01[r], __ATOMIC_RELAXED, __HIP_MEMORY_SCOPE_AGENT);
        }
        const int erow1 = e0 + 16 + r + 8 * hi;
        if (erow1 < E) {
            const int dst = ei[E + erow1];
            float* o = out + (size_t)dst * 32 + m;
            __hip_atomic_fetch_add(o,      c10[r], __ATOMIC_RELAXED, __HIP_MEMORY_SCOPE_AGENT);
            __hip_atomic_fetch_add(o + 16, c11[r], __ATOMIC_RELAXED, __HIP_MEMORY_SCOPE_AGENT);
        }
    }
}

// ======================= root transform (out init) =========================
// out[n] = x[n] @ root + bias   (writes every element -> runs before atomics)
__global__ __launch_bounds__(256) void nnconv_root_kernel(
    const float* __restrict__ x,      // [N, 32]
    const float* __restrict__ root,   // [32, 32]
    const float* __restrict__ bias,   // [32]
    float*       __restrict__ out,    // [N, 32]
    int N)
{
    __shared__ __align__(32) _Float16 ldsB[2 * 512];
    const int t = threadIdx.x;
    for (int i = t; i < 1024; i += 256) {
        const int frag = i >> 9;
        const int rem  = i & 511;
        const int l    = rem >> 4;
        const int j    = rem & 15;
        const int hi   = (l >> 4) & 1;
        const int k    = (j >> 3) * 16 + hi * 8 + (j & 7);
        const int g    = frag * 16 + (l & 15);
        ldsB[i] = (_Float16)root[k * 32 + g];
    }
    __syncthreads();

    const int lane = t & 31;
    const int wave = t >> 5;
    const int m    = lane & 15;
    const int hi   = lane >> 4;
    const bool hiF = hi != 0;

    const int n0 = blockIdx.x * 128 + wave * 16;
    int node = n0 + m;
    if (node > N - 1) node = N - 1;

    // x row -> 16 h2 pairs, then A slots: 0-7: f=hi*8+j, 8-15: f=16+hi*8+j
    const v4f* xp4 = (const v4f*)(x + (size_t)node * 32);
    h2 xh2[16];
    #pragma unroll
    for (int q = 0; q < 8; ++q) {
        const v4f xq = xp4[q];
        xh2[2 * q]     = h2{(_Float16)xq[0], (_Float16)xq[1]};
        xh2[2 * q + 1] = h2{(_Float16)xq[2], (_Float16)xq[3]};
    }
    AFrag a;
    #pragma unroll
    for (int k = 0; k < 4; ++k) {
        a.p[k]     = hiF ? xh2[4 + k]  : xh2[k];
        a.p[4 + k] = hiF ? xh2[12 + k] : xh2[8 + k];
    }

    const float b0v = bias[m];
    const float b1v = bias[16 + m];
    v8f c0, c1;
    #pragma unroll
    for (int r = 0; r < 8; ++r) { c0[r] = b0v; c1[r] = b1v; }   // bias in accumulator

    const v16h bf0 = *(const v16h*)&ldsB[lane * 16];
    const v16h bf1 = *(const v16h*)&ldsB[512 + lane * 16];
    c0 = __builtin_amdgcn_wmma_f32_16x16x32_f16(false, a.v, false, bf0, (short)0, c0, false, false);
    c1 = __builtin_amdgcn_wmma_f32_16x16x32_f16(false, a.v, false, bf1, (short)0, c1, false, false);

    #pragma unroll
    for (int r = 0; r < 8; ++r) {
        const int nrow = n0 + r + 8 * hi;
        if (nrow < N) {
            out[(size_t)nrow * 32 + m]      = c0[r];
            out[(size_t)nrow * 32 + 16 + m] = c1[r];
        }
    }
}

// ============================== launch =====================================
extern "C" void kernel_launch(void* const* d_in, const int* in_sizes, int n_in,
                              void* d_out, int out_size, void* d_ws, size_t ws_size,
                              hipStream_t stream) {
    const float* x_a      = (const float*)d_in[0];
    const float* x_b      = (const float*)d_in[1];
    const int*   ei_ab    = (const int*)  d_in[2];
    const float* ea_ab    = (const float*)d_in[3];
    const int*   ei_ba    = (const int*)  d_in[4];
    const float* ea_ba    = (const float*)d_in[5];
    const float* msg_W_ab = (const float*)d_in[6];
    const float* msg_b_ab = (const float*)d_in[7];
    const float* root_ab  = (const float*)d_in[8];
    const float* bias_ab  = (const float*)d_in[9];
    const float* msg_W_ba = (const float*)d_in[10];
    const float* msg_b_ba = (const float*)d_in[11];
    const float* root_ba  = (const float*)d_in[12];
    const float* bias_ba  = (const float*)d_in[13];

    const int N_A  = in_sizes[0] / 32;
    const int N_B  = in_sizes[1] / 32;
    const int E_ab = in_sizes[2] / 2;
    const int E_ba = in_sizes[4] / 2;

    float* out   = (float*)d_out;
    float* out_a = out;                       // dst type a  (conv b->a)
    float* out_b = out + (size_t)N_A * 32;    // dst type b  (conv a->b)

    // 1) Initialize outputs with root transform + bias (covers every element).
    nnconv_root_kernel<<<(N_A + 127) / 128, 256, 0, stream>>>(x_a, root_ba, bias_ba, out_a, N_A);
    nnconv_root_kernel<<<(N_B + 127) / 128, 256, 0, stream>>>(x_b, root_ab, bias_ab, out_b, N_B);

    // 2) Fused edge-conditioned message GEMM + atomic segment-sum scatter.
    nnconv_edge_kernel<<<(E_ab + 255) / 256, 256, 0, stream>>>(
        x_a, ei_ab, ea_ab, msg_W_ab, msg_b_ab, out_b, E_ab);
    nnconv_edge_kernel<<<(E_ba + 255) / 256, 256, 0, stream>>>(
        x_b, ei_ba, ea_ba, msg_W_ba, msg_b_ba, out_a, E_ba);
}
